// WeightMap_65953517798002
// MI455X (gfx1250) — compile-verified, matrix operand order checked by reading, not run on hardware
//
#include <hip/hip_runtime.h>
#include <hip/hip_bf16.h>
#include <math.h>

typedef __attribute__((ext_vector_type(16))) _Float16 v16h;
typedef __attribute__((ext_vector_type(8)))  float    v8f;

#define BD   16
#define CD   4
#define HD   512
#define WD   512
#define SLAB 32768          // 16*4*512 elements per (s,i) EDT slab
#define NSI  8              // 2 samplings * 4 EDT rows
#define BIGF 1.0e10f

// ---------------------------------------------------------------------------
// Build the 8 squared-EDT seed slabs g1[s][i][b][c][w] from x[b][c][i][w],
// and zero the 4-float channel-sum accumulator.
// ---------------------------------------------------------------------------
__global__ void k_init(const float* __restrict__ x, float* __restrict__ g1,
                       float* __restrict__ wsum) {
  int tid = blockIdx.x * blockDim.x + threadIdx.x;   // [0, 8*32768)
  if (tid < 4) wsum[tid] = 0.0f;
  int w = tid & 511;
  int c = (tid >> 9) & 3;
  int b = (tid >> 11) & 15;
  int i = (tid >> 15) & 3;                           // s = tid>>17 (x-independent)
  float xv = x[(((b * CD + c) * HD + i) * WD) + w];
  g1[tid] = (xv > 0.0f) ? BIGF : 0.0f;
}

// ---------------------------------------------------------------------------
// Channel sums via V_WMMA_F32_16X16X32_F16: each wave packs 512 floats of one
// channel slab into a 16x32 f16 A-matrix (0/1 exact in f16), multiplies by an
// all-ones B (D rows = row sums, f32 accumulation, exact integers), then
// reduces lanes + waves and does one f32 atomic per block.
// ---------------------------------------------------------------------------
__global__ void k_chansum(const float* __restrict__ x, float* __restrict__ wsum) {
  __shared__ float wpart[8];
  int wave = threadIdx.x >> 5;
  int lane = threadIdx.x & 31;
  long long base = (((long long)blockIdx.x * 8 + wave) << 9) + ((long long)lane << 4);
  const float4* p4 = reinterpret_cast<const float4*>(x + base);
  v16h a;
#pragma unroll
  for (int j = 0; j < 4; ++j) {
    float4 f = p4[j];
    a[4 * j + 0] = (_Float16)f.x;
    a[4 * j + 1] = (_Float16)f.y;
    a[4 * j + 2] = (_Float16)f.z;
    a[4 * j + 3] = (_Float16)f.w;
  }
  v16h bones;
#pragma unroll
  for (int j = 0; j < 16; ++j) bones[j] = (_Float16)1.0f;
  v8f acc = {};
  acc = __builtin_amdgcn_wmma_f32_16x16x32_f16(
      /*neg_a=*/false, a, /*neg_b=*/false, bones,
      /*c_mod=*/(short)0, acc, /*reuse_a=*/false, /*reuse_b=*/false);
  float part = 0.0f;
#pragma unroll
  for (int j = 0; j < 8; ++j) part += acc[j];
#pragma unroll
  for (int off = 16; off > 0; off >>= 1)
    part += __shfl_xor(part, off, 32);               // sums all 256 D elems = 16*total
  if (lane == 0) wpart[wave] = part;
  __syncthreads();
  if (threadIdx.x == 0) {
    float tot = 0.0f;
#pragma unroll
    for (int j = 0; j < 8; ++j) tot += wpart[j];
    // block spans 4096 contiguous elements inside one (b,c) slab of 2^18
    int c = (int)((((long long)blockIdx.x << 12) >> 18) & 3);
    atomicAdd(wsum + c, tot * (1.0f / 16.0f));
  }
}

// ---------------------------------------------------------------------------
// Min-plus pass over the B axis (n=16).
// ---------------------------------------------------------------------------
__global__ void k_pass_b(const float* __restrict__ gin, float* __restrict__ gout) {
  int tid = blockIdx.x * blockDim.x + threadIdx.x;
  int w  = tid & 511;
  int c  = (tid >> 9) & 3;
  int b  = (tid >> 11) & 15;
  int si = tid >> 15;                                // s*4 + i
  float samp = (si >= 4) ? 2.0f : 1.0f;
  int rowbase = si * SLAB + c * 512 + w;
  float m = 3.4e38f;
#pragma unroll
  for (int q = 0; q < 16; ++q) {
    float d = samp * (float)(b - q);
    m = fminf(m, gin[rowbase + q * 2048] + d * d);
  }
  gout[tid] = m;
}

// ---------------------------------------------------------------------------
// Min-plus pass over the C axis (n=4).
// ---------------------------------------------------------------------------
__global__ void k_pass_c(const float* __restrict__ gin, float* __restrict__ gout) {
  int tid = blockIdx.x * blockDim.x + threadIdx.x;
  int w  = tid & 511;
  int c  = (tid >> 9) & 3;
  int b  = (tid >> 11) & 15;
  int si = tid >> 15;
  float samp = (si >= 4) ? 2.0f : 1.0f;
  int rowbase = si * SLAB + b * 2048 + w;
  float m = 3.4e38f;
#pragma unroll
  for (int q = 0; q < 4; ++q) {
    float d = samp * (float)(c - q);
    m = fminf(m, gin[rowbase + q * 512] + d * d);
  }
  gout[tid] = m;
}

// ---------------------------------------------------------------------------
// Min-plus pass over the W axis (n=512), staged in LDS for both samplings,
// fused with sqrt + gaussian weighting into dbuf[i][b][c][w].
// ---------------------------------------------------------------------------
__global__ void k_pass_w(const float* __restrict__ gin, float* __restrict__ dbuf) {
  __shared__ float row[2][WD];
  int blk = blockIdx.x;                              // i*64 + b*4 + c
  int c = blk & 3;
  int b = (blk >> 2) & 15;
  int i = blk >> 6;
  int p = threadIdx.x;                               // 512 threads
  row[0][p] = gin[(0 * 4 + i) * SLAB + b * 2048 + c * 512 + p];
  row[1][p] = gin[(1 * 4 + i) * SLAB + b * 2048 + c * 512 + p];
  __syncthreads();
  float dsum = 0.0f;
#pragma unroll
  for (int s = 0; s < 2; ++s) {
    float samp = (s == 0) ? 1.0f : 2.0f;
    float m = 3.4e38f;
    for (int q = 0; q < WD; ++q) {
      float d = samp * (float)(p - q);
      m = fminf(m, row[s][q] + d * d);
    }
    dsum += sqrtf(m);
  }
  float t = dsum * (1.0f / 5.0f);                    // (d1+d2)/SIGMA
  dbuf[i * SLAB + b * 2048 + c * 512 + p] = 10.0f * expf(-0.5f * t * t);
}

// ---------------------------------------------------------------------------
// Final memory-bound write: out = w[c] + (h<4 ? dbuf : 10), float4 stores.
// ---------------------------------------------------------------------------
__global__ void k_final(const float* __restrict__ wsum, const float* __restrict__ dbuf,
                        float* __restrict__ out) {
  long long t = (long long)blockIdx.x * blockDim.x + threadIdx.x;  // group of 4
  long long e = t << 2;
  int w = (int)(e & 511);
  int h = (int)((e >> 9) & 511);
  int c = (int)((e >> 18) & 3);
  int b = (int)(e >> 20);
  float wc = wsum[c] * (1.0f / 16777216.0f);
  float4 v;
  if (h < 4) {
    float4 d = *reinterpret_cast<const float4*>(dbuf + h * SLAB + b * 2048 + c * 512 + w);
    v = make_float4(wc + d.x, wc + d.y, wc + d.z, wc + d.w);
  } else {
    float s = wc + 10.0f;
    v = make_float4(s, s, s, s);
  }
  reinterpret_cast<float4*>(out)[t] = v;
}

// ---------------------------------------------------------------------------
extern "C" void kernel_launch(void* const* d_in, const int* in_sizes, int n_in,
                              void* d_out, int out_size, void* d_ws, size_t ws_size,
                              hipStream_t stream) {
  const float* x = (const float*)d_in[0];
  float* out = (float*)d_out;
  float* ws  = (float*)d_ws;

  float* wsum = ws;                    // 4 floats (64-float slot for alignment)
  float* g1   = ws + 64;               // 8 * 32768 floats
  float* g2   = g1 + NSI * SLAB;       // 8 * 32768 floats
  float* dbuf = g2 + NSI * SLAB;       // 4 * 32768 floats
  // total scratch: (64 + 2*262144 + 131072) * 4 B ≈ 2.62 MB (L2-resident)

  k_init   <<<NSI * SLAB / 256, 256, 0, stream>>>(x, g1, wsum);
  k_chansum<<<4096,            256, 0, stream>>>(x, wsum);
  k_pass_b <<<NSI * SLAB / 256, 256, 0, stream>>>(g1, g2);
  k_pass_c <<<NSI * SLAB / 256, 256, 0, stream>>>(g2, g1);
  k_pass_w <<<256,             512, 0, stream>>>(g1, dbuf);
  k_final  <<<16384,           256, 0, stream>>>(wsum, dbuf, out);
}
